// RelativePartialMultiHeadSelfAttention_5772436046473
// MI455X (gfx1250) — compile-verified, matrix operand order checked by reading, not run on hardware
//
#include <hip/hip_runtime.h>
#include <hip/hip_bf16.h>

// ---------------------------------------------------------------------------
// Transformer-XL relative multi-head attention for MI455X (gfx1250).
// All GEMMs run on v_wmma_f32_16x16x32_f16 (wave32 WMMA, f32 accumulate).
// Attention is fused flash-style (online softmax) so the (B,H,Q,KLEN) score
// matrix (256MB f32) never touches HBM. The relative-shift is folded into the
// flash loop via a 47-wide diagonal band of qr*r^T tiles gathered through LDS.
// Workspace requirement: ~62 MB of f16 scratch in d_ws.
// ---------------------------------------------------------------------------

enum { B_ = 2, Q_ = 1024, MEM_ = 1024, U_ = 1024, H_ = 16, DH_ = 64, KLEN_ = MEM_ + Q_ };

typedef __attribute__((ext_vector_type(16))) _Float16 v16h;
typedef __attribute__((ext_vector_type(8)))  _Float16 v8h;
typedef __attribute__((ext_vector_type(8)))  float    v8f;

// ---- WMMA fragment helpers (layouts per CDNA5 ISA 7.12.2) -----------------
// A fragment 16x32 f16: lane m=lane&15 holds row m; half=lane>>4 selects
// k-groups: regs 0-3 = k in [half*8, half*8+8), regs 4-7 = +16. Each group is
// 8 contiguous halves -> one 16B load. B fragment (NT form, B stored as NxK
// row-major) is symmetric with N in place of M.
__device__ __forceinline__ v16h frag_combine(v8h lo, v8h hi) {
  return __builtin_shufflevector(lo, hi, 0,1,2,3,4,5,6,7,8,9,10,11,12,13,14,15);
}

__device__ __forceinline__ v16h ldfrag(const _Float16* base, int ld, int koff) {
  int lane = threadIdx.x & 31;
  int m = lane & 15, hf = lane >> 4;
  const v8h* p = (const v8h*)(base + (size_t)m * ld + koff + hf * 8);
  return frag_combine(p[0], p[2]);   // p[2] == +16 halves
}

__device__ __forceinline__ v16h ldfrag_rowclamp(const _Float16* base, int ld,
                                                int row0, int maxrow, int koff) {
  int lane = threadIdx.x & 31;
  int m = lane & 15, hf = lane >> 4;
  int row = row0 + m; if (row > maxrow) row = maxrow;
  const v8h* p = (const v8h*)(base + (size_t)row * ld + koff + hf * 8);
  return frag_combine(p[0], p[2]);
}

__device__ __forceinline__ v8f wmma_(v16h a, v16h b, v8f c) {
  return __builtin_amdgcn_wmma_f32_16x16x32_f16(false, a, false, b, (short)0, c,
                                                false, false);
}

// ---- pack / convert kernels ----------------------------------------------
__global__ void pack_concat_f16(const float* __restrict__ mem,
                                const float* __restrict__ inp,
                                _Float16* __restrict__ xfull) {
  size_t n = (size_t)B_ * KLEN_ * U_;
  for (size_t idx = (size_t)blockIdx.x * blockDim.x + threadIdx.x; idx < n;
       idx += (size_t)gridDim.x * blockDim.x) {
    size_t b = idx / ((size_t)KLEN_ * U_);
    size_t rem = idx - b * (size_t)KLEN_ * U_;
    int row = (int)(rem / U_), col = (int)(rem % U_);
    float v = (row < MEM_) ? mem[(b * MEM_ + row) * U_ + col]
                           : inp[(b * Q_ + (row - MEM_)) * U_ + col];
    xfull[idx] = (_Float16)v;
  }
}

__global__ void cvt_f32_to_f16(const float* __restrict__ in,
                               _Float16* __restrict__ out, size_t n) {
  for (size_t idx = (size_t)blockIdx.x * blockDim.x + threadIdx.x; idx < n;
       idx += (size_t)gridDim.x * blockDim.x)
    out[idx] = (_Float16)in[idx];
}

// out[n*K + k] = in[k*ldin + coloff + n]  (weight transpose for NT GEMMs)
__global__ void transpose_w_f16(const float* __restrict__ in,
                                _Float16* __restrict__ out,
                                int K, int N, int ldin, int coloff) {
  size_t n = (size_t)K * N;
  for (size_t idx = (size_t)blockIdx.x * blockDim.x + threadIdx.x; idx < n;
       idx += (size_t)gridDim.x * blockDim.x) {
    int nn = (int)(idx / K), k = (int)(idx % K);
    out[idx] = (_Float16)in[(size_t)k * ldin + coloff + nn];
  }
}

// ---- generic batched NT GEMM: C(MxN) = A(MxK) * Bw(NxK)^T ------------------
// 256 threads = 8 waves, wave grid 2x4 -> 32x64 block tile, 16x16 per wave.
enum { GM_PLAIN = 0, GM_QCQR = 1, GM_VT = 2, GM_F32 = 3 };

__global__ __launch_bounds__(256)
void gemm_nt(const _Float16* __restrict__ A, size_t strideA, int lda,
             const _Float16* __restrict__ Bw, int ldb, int K,
             _Float16* __restrict__ out0, size_t strideO, int ldo,
             _Float16* __restrict__ out1,
             const float* __restrict__ bias0, const float* __restrict__ bias1,
             float* __restrict__ outf, size_t strideF,
             int klen_vt, int mode) {
  int b   = blockIdx.z;
  int wid = threadIdx.x >> 5;
  int m0  = blockIdx.y * 32 + (wid >> 2) * 16;
  int n0  = blockIdx.x * 64 + (wid & 3) * 16;
  const _Float16* Ab = A + (size_t)b * strideA + (size_t)m0 * lda;
  const _Float16* Bb = Bw + (size_t)n0 * ldb;

  v8f c = {};
  for (int k0 = 0; k0 < K; k0 += 32)
    c = wmma_(ldfrag(Ab, lda, k0), ldfrag(Bb, ldb, k0), c);

  int lane = threadIdx.x & 31, jl = lane & 15, hf = lane >> 4;
#pragma unroll
  for (int r = 0; r < 8; r++) {
    int row = m0 + r + 8 * hf;
    int col = n0 + jl;
    float v = c[r];
    if (mode == GM_QCQR) {
      size_t o = (size_t)b * strideO + (size_t)row * ldo + col;
      out0[o] = (_Float16)(v + bias0[col]);
      out1[o] = (_Float16)(v + bias1[col]);
    } else if (mode == GM_VT) {           // store V transposed: (B,H,DH,KLEN)
      int h = col >> 6, d = col & 63;
      out0[(((size_t)b * H_ + h) * DH_ + d) * (size_t)klen_vt + row] = (_Float16)v;
    } else if (mode == GM_F32) {
      outf[(size_t)b * strideF + (size_t)row * ldo + col] = v;
    } else {
      out0[(size_t)b * strideO + (size_t)row * ldo + col] = (_Float16)v;
    }
  }
}

// ---- fused flash attention with relative shift ----------------------------
// grid (Q/64, H, B), 128 threads = 4 independent waves; each wave owns a
// 16-row q tile and streams the key dimension in 32-wide tiles with online
// softmax. rel_shift(qr*r^T)[i,j] = S_r[i, j+Q-1-i]; the needed r-band per
// (i-tile, j-tile) is 47 wide -> three 16-col WMMA tiles gathered via LDS.
// The causal mask bound (j <= MEM+i) exactly covers the shift's OOB region.
__global__ __launch_bounds__(128)
void flash_attn(const _Float16* __restrict__ qc, const _Float16* __restrict__ qr,
                const _Float16* __restrict__ kk, const _Float16* __restrict__ rr,
                const _Float16* __restrict__ vT, _Float16* __restrict__ outp) {
  __shared__ __align__(16) float    relbuf[4][16][49];
  __shared__ __align__(16) _Float16 pbuf[4][16][32];

  int w = threadIdx.x >> 5;
  int lane = threadIdx.x & 31, jl = lane & 15, hf = lane >> 4;
  int b = blockIdx.z, h = blockIdx.y;
  int i0 = blockIdx.x * 64 + w * 16;

  const _Float16* qcb = qc + ((size_t)b * Q_ + i0) * U_ + h * DH_;
  const _Float16* qrb = qr + ((size_t)b * Q_ + i0) * U_ + h * DH_;
  v16h aqc0 = ldfrag(qcb, U_, 0), aqc1 = ldfrag(qcb, U_, 32);
  v16h aqr0 = ldfrag(qrb, U_, 0), aqr1 = ldfrag(qrb, U_, 32);

  const _Float16* kbase = kk + (size_t)b * KLEN_ * U_ + h * DH_;
  const _Float16* rbase = rr + (size_t)b * KLEN_ * U_ + h * DH_;

  v8f o[4] = {};
  float mrow[8], lrow[8];
#pragma unroll
  for (int r = 0; r < 8; r++) { mrow[r] = -1e30f; lrow[r] = 0.f; }

  int jmax = MEM_ + i0 + 15;                 // causal bound, halves the work
  for (int j0 = 0; j0 <= jmax; j0 += 32) {
    // context scores: two 16-col subtiles over K=DH=64
    v8f sc0 = {}, sc1 = {};
    sc0 = wmma_(aqc0, ldfrag(kbase + (size_t)j0 * U_, U_, 0), sc0);
    sc0 = wmma_(aqc1, ldfrag(kbase + (size_t)j0 * U_, U_, 32), sc0);
    sc1 = wmma_(aqc0, ldfrag(kbase + (size_t)(j0 + 16) * U_, U_, 0), sc1);
    sc1 = wmma_(aqc1, ldfrag(kbase + (size_t)(j0 + 16) * U_, U_, 32), sc1);

    // relative-score band: r rows rb0 .. rb0+47 (rows past KLEN are masked)
    int rb0 = j0 + (Q_ - 16) - i0;
#pragma unroll
    for (int tt = 0; tt < 3; tt++) {
      v8f rc = {};
      rc = wmma_(aqr0, ldfrag_rowclamp(rbase, U_, rb0 + tt * 16, KLEN_ - 1, 0), rc);
      rc = wmma_(aqr1, ldfrag_rowclamp(rbase, U_, rb0 + tt * 16, KLEN_ - 1, 32), rc);
#pragma unroll
      for (int r = 0; r < 8; r++)
        relbuf[w][r + 8 * hf][tt * 16 + jl] = rc[r];
    }
    asm volatile("" ::: "memory");   // order LDS write -> gather (same wave)

    // gather shifted rel term, scale 1/sqrt(64), apply causal mask
    float s0[8], s1[8];
#pragma unroll
    for (int r = 0; r < 8; r++) {
      int iloc = r + 8 * hf;
      float rel0 = relbuf[w][iloc][jl + 15 - iloc];
      float rel1 = relbuf[w][iloc][16 + jl + 15 - iloc];
      int lim = MEM_ + i0 + iloc;
      s0[r] = (j0 + jl      <= lim) ? (sc0[r] + rel0) * 0.125f : -1e30f;
      s1[r] = (j0 + 16 + jl <= lim) ? (sc1[r] + rel1) * 0.125f : -1e30f;
    }

    // online softmax (row = reg r + 8*hf; 16-lane group holds the 16 cols)
    float alpha[8];
#pragma unroll
    for (int r = 0; r < 8; r++) {
      float tm = fmaxf(s0[r], s1[r]);
#pragma unroll
      for (int off = 1; off < 16; off <<= 1) tm = fmaxf(tm, __shfl_xor(tm, off, 32));
      float mn = fmaxf(mrow[r], tm);
      alpha[r] = __expf(mrow[r] - mn);
      float p0 = __expf(s0[r] - mn), p1 = __expf(s1[r] - mn);
      float ps = p0 + p1;
#pragma unroll
      for (int off = 1; off < 16; off <<= 1) ps += __shfl_xor(ps, off, 32);
      lrow[r] = lrow[r] * alpha[r] + ps;
      mrow[r] = mn;
      int iloc = r + 8 * hf;
      pbuf[w][iloc][jl]      = (_Float16)p0;
      pbuf[w][iloc][16 + jl] = (_Float16)p1;
    }
#pragma unroll
    for (int dt = 0; dt < 4; dt++)
#pragma unroll
      for (int r = 0; r < 8; r++) o[dt][r] *= alpha[r];

    asm volatile("" ::: "memory");   // order P store -> A-fragment reload
    v16h ap = ldfrag(&pbuf[w][0][0], 32, 0);   // C-layout -> A-layout via LDS
#pragma unroll
    for (int dt = 0; dt < 4; dt++) {
      const _Float16* vb = vT + (((size_t)b * H_ + h) * DH_ + dt * 16) * KLEN_;
      o[dt] = wmma_(ap, ldfrag(vb, KLEN_, j0), o[dt]);
    }
  }

  // normalize and store (B,Q,U) f16
#pragma unroll
  for (int dt = 0; dt < 4; dt++)
#pragma unroll
    for (int r = 0; r < 8; r++) {
      int iloc = r + 8 * hf;
      float vv = o[dt][r] / lrow[r];
      outp[((size_t)b * Q_ + i0 + iloc) * U_ + h * DH_ + dt * 16 + jl] = (_Float16)vv;
    }
}

// ---------------------------------------------------------------------------
extern "C" void kernel_launch(void* const* d_in, const int* in_sizes, int n_in,
                              void* d_out, int out_size, void* d_ws, size_t ws_size,
                              hipStream_t stream) {
  (void)in_sizes; (void)n_in; (void)out_size; (void)ws_size;
  const float* inputs    = (const float*)d_in[0];
  const float* relatives = (const float*)d_in[1];
  const float* memories  = (const float*)d_in[2];
  const float* biasc     = (const float*)d_in[3];
  const float* biasr     = (const float*)d_in[4];
  const float* Wq        = (const float*)d_in[5];
  const float* Wkv       = (const float*)d_in[6];
  const float* Wr        = (const float*)d_in[7];
  const float* Wo        = (const float*)d_in[8];
  float* out = (float*)d_out;

  // f16 workspace carve-out (~62 MB)
  _Float16* ws = (_Float16*)d_ws;
  size_t off = 0;
  auto carve = [&](size_t n) { _Float16* p = ws + off; off += n; return p; };
  const size_t BKU = (size_t)B_ * KLEN_ * U_;   // 4M
  const size_t BQU = (size_t)B_ * Q_ * U_;      // 2M
  const size_t UU  = (size_t)U_ * U_;           // 1M
  _Float16* xfull  = carve(BKU);
  _Float16* rel16  = carve(BKU);
  _Float16* WqT    = carve(UU);
  _Float16* WkT    = carve(UU);
  _Float16* WvT    = carve(UU);
  _Float16* WrT    = carve(UU);
  _Float16* WoT    = carve(UU);
  _Float16* qcb    = carve(BQU);
  _Float16* qrb    = carve(BQU);
  _Float16* kbuf   = carve(BKU);
  _Float16* rbuf   = carve(BKU);
  _Float16* vTb    = carve(BKU);                // (B,H,DH,KLEN)
  _Float16* attout = carve(BQU);

  // 1) pack + convert
  pack_concat_f16<<<4096, 256, 0, stream>>>(memories, inputs, xfull);
  cvt_f32_to_f16<<<4096, 256, 0, stream>>>(relatives, rel16, BKU);
  transpose_w_f16<<<2048, 256, 0, stream>>>(Wq,  WqT, U_, U_, U_,     0);
  transpose_w_f16<<<2048, 256, 0, stream>>>(Wkv, WkT, U_, U_, 2 * U_, 0);
  transpose_w_f16<<<2048, 256, 0, stream>>>(Wkv, WvT, U_, U_, 2 * U_, U_);
  transpose_w_f16<<<2048, 256, 0, stream>>>(Wr,  WrT, U_, U_, U_,     0);
  transpose_w_f16<<<2048, 256, 0, stream>>>(Wo,  WoT, U_, U_, U_,     0);

  // 2) projections (all WMMA NT GEMMs)
  dim3 gq(U_ / 64, Q_ / 32, B_);
  gemm_nt<<<gq, 256, 0, stream>>>(xfull + (size_t)MEM_ * U_, BKU / B_, U_,
                                  WqT, U_, U_,
                                  qcb, BQU / B_, U_, qrb, biasc, biasr,
                                  nullptr, 0, 0, GM_QCQR);
  dim3 gk(U_ / 64, KLEN_ / 32, B_);
  gemm_nt<<<gk, 256, 0, stream>>>(xfull, BKU / B_, U_, WkT, U_, U_,
                                  kbuf, BKU / B_, U_, nullptr, nullptr, nullptr,
                                  nullptr, 0, 0, GM_PLAIN);
  gemm_nt<<<gk, 256, 0, stream>>>(rel16, BKU / B_, U_, WrT, U_, U_,
                                  rbuf, BKU / B_, U_, nullptr, nullptr, nullptr,
                                  nullptr, 0, 0, GM_PLAIN);
  gemm_nt<<<gk, 256, 0, stream>>>(xfull, BKU / B_, U_, WvT, U_, U_,
                                  vTb, 0, U_, nullptr, nullptr, nullptr,
                                  nullptr, 0, KLEN_, GM_VT);

  // 3) fused attention (no score matrix in HBM)
  flash_attn<<<dim3(Q_ / 64, H_, B_), 128, 0, stream>>>(qcb, qrb, kbuf, rbuf,
                                                        vTb, attout);

  // 4) output projection, f32 result
  gemm_nt<<<gq, 256, 0, stream>>>(attout, BQU / B_, U_, WoT, U_, U_,
                                  nullptr, 0, U_, nullptr, nullptr, nullptr,
                                  out, BQU / B_, 0, GM_F32);
}
